// Attention_20151986553284
// MI455X (gfx1250) — matrix operand order repaired and checked by reading, not compiled, not run on hardware
//
#include <hip/hip_runtime.h>
#include <hip/hip_bf16.h>

// ---------------------------------------------------------------------------
// GPT-2 attention block for MI455X (gfx1250, wave32, WMMA f16 -> f32 accum).
//   qkv  = x @ w_attn + b_attn          (WMMA GEMM, f16 in, f16 out)
//   attn = flash(q,k,v) w/ mask=1e-9    (WMMA scores + online softmax + WMMA PV)
//   out  = attn @ w_proj + b_proj       (WMMA GEMM, f16 in, f32 out)
// Data movement: double-buffered LDS tiles staged with GLOBAL_LOAD_ASYNC_TO_LDS
// (ASYNCcnt) when available, manual copies otherwise.
// ---------------------------------------------------------------------------

typedef __attribute__((ext_vector_type(16))) _Float16 v16h;
typedef __attribute__((ext_vector_type(8)))  float    v8f;
typedef int v4i32 __attribute__((vector_size(16)));   // matches builtin proto

union V16U { v16h h; uint4 u[2]; };

#define AS1G __attribute__((address_space(1)))
#define AS3L __attribute__((address_space(3)))

#if defined(__has_builtin)
#if __has_builtin(__builtin_amdgcn_global_load_async_to_lds_b128)
#define ASYNC_LDS 1
#endif
#endif
#ifndef ASYNC_LDS
#define ASYNC_LDS 0
#endif

// copy 16B global -> LDS (per lane); async DMA when available
static __device__ __forceinline__ void cp16(const _Float16* g, _Float16* l) {
#if ASYNC_LDS
  __builtin_amdgcn_global_load_async_to_lds_b128(
      (AS1G v4i32*)(uintptr_t)g, (AS3L v4i32*)(uintptr_t)l,
      /*offset=*/0, /*cpol=*/0);
#else
  *(uint4*)l = *(const uint4*)g;
#endif
}

static __device__ __forceinline__ void cp_wait() {
#if ASYNC_LDS
#if __has_builtin(__builtin_amdgcn_s_wait_asynccnt)
  __builtin_amdgcn_s_wait_asynccnt(0);
#else
  asm volatile("s_wait_asynccnt 0x0" ::: "memory");
#endif
#endif
}

static __device__ __forceinline__ v8f wmma16(v16h a, v16h b, v8f c) {
  // D = A(16x32 f16) x B(32x16 f16) + C(16x16 f32)
  return __builtin_amdgcn_wmma_f32_16x16x32_f16(
      /*neg_a=*/false, a, /*neg_b=*/false, b,
      /*c_mod=*/(short)0, c, /*reuse_a=*/false, /*reuse_b=*/false);
}

// ------------------------------ convert ------------------------------------
__global__ __launch_bounds__(256) void cvt_f32_f16(const float* __restrict__ in,
                                                   _Float16* __restrict__ out,
                                                   int n) {
  int i = blockIdx.x * blockDim.x + threadIdx.x;
  if (i < n) out[i] = (_Float16)in[i];
}

// ------------------------------ GEMM ---------------------------------------
// C[M,N] = A[M,K](f16) @ W[K,N](f16) + bias[N](f32), OutT in {float,_Float16}.
// Block: 256 thr = 8 waves (2 M-waves x 4 N-waves). Block tile 64M x 128N.
// Double-buffered LDS staging (A tile async-DMA, W tile transposed manually).
template <typename OutT>
__global__ __launch_bounds__(256) void gemm_wmma_kernel(
    const _Float16* __restrict__ A, const _Float16* __restrict__ W,
    const float* __restrict__ bias, OutT* __restrict__ C,
    int M, int N, int K) {
  constexpr int ASTR = 40;  // At row stride (halfs): 80B rows keep 16B align
  constexpr int BSTR = 40;  // Bt row stride (halfs)
  __shared__ _Float16 At[2][64 * ASTR];    // [m][k]  64x32 tile
  __shared__ _Float16 Bt[2][128 * BSTR];   // [n][k]  32x128 tile, transposed

  const int tid  = threadIdx.x;
  const int wid  = tid >> 5;
  const int lane = tid & 31;
  const int lo   = lane & 15;
  const int hi   = lane >> 4;           // 0 or 1
  const int m0 = blockIdx.y * 64;
  const int n0 = blockIdx.x * 128;
  const int wm = (wid >> 2) * 32;       // wave M offset in block tile
  const int wn = (wid & 3) * 32;        // wave N offset in block tile

  v8f acc[2][2] = {};

  auto stage = [&](int k0, int buf) {
    // A 64x32: one b128 per thread, straight copy -> async DMA
    {
      int row = tid >> 2, ca = (tid & 3) * 8;
      cp16(A + (size_t)(m0 + row) * K + k0 + ca, &At[buf][row * ASTR + ca]);
    }
    // W 32x128 transposed into Bt[n][k] (needs VGPR bounce)
    for (int c = tid; c < 512; c += 256) {
      int kr = c >> 4, nc = (c & 15) * 8;
      uint4 u = *(const uint4*)(W + (size_t)(k0 + kr) * N + n0 + nc);
      const _Float16* hs = (const _Float16*)&u;
#pragma unroll
      for (int e = 0; e < 8; ++e) Bt[buf][(nc + e) * BSTR + kr] = hs[e];
    }
  };

  stage(0, 0);
  int cur = 0;
  for (int k0 = 0; k0 < K; k0 += 32) {
    cp_wait();           // our async stores for tile k0 have landed in LDS
    __syncthreads();     // everyone's tile k0 visible; prev compute finished
    if (k0 + 32 < K) stage(k0 + 32, cur ^ 1);  // overlap next tile with WMMA

    V16U af[2], bf[2];
#pragma unroll
    for (int t = 0; t < 2; ++t) {
      // A frag: lane=M, element i -> K = hi*16 + i
      const _Float16* ar = &At[cur][(wm + t * 16 + lo) * ASTR + hi * 16];
      af[t].u[0] = *(const uint4*)(ar);
      af[t].u[1] = *(const uint4*)(ar + 8);
      // B frag: lane=N, element i -> K = hi*16 + i (contiguous in Bt)
      const _Float16* br = &Bt[cur][(wn + t * 16 + lo) * BSTR + hi * 16];
      bf[t].u[0] = *(const uint4*)(br);
      bf[t].u[1] = *(const uint4*)(br + 8);
    }
#pragma unroll
    for (int tm = 0; tm < 2; ++tm)
#pragma unroll
      for (int tn = 0; tn < 2; ++tn)
        acc[tm][tn] = wmma16(af[tm].h, bf[tn].h, acc[tm][tn]);
    cur ^= 1;
  }

  // epilogue: bias + store. C layout: VGPR r -> row r + hi*8, N = lo.
#pragma unroll
  for (int tn = 0; tn < 2; ++tn) {
    int ng = n0 + wn + tn * 16 + lo;
    float bv = bias[ng];
#pragma unroll
    for (int tm = 0; tm < 2; ++tm)
#pragma unroll
      for (int r = 0; r < 8; ++r) {
        int row = m0 + wm + tm * 16 + hi * 8 + r;
        C[(size_t)row * N + ng] = (OutT)(acc[tm][tn][r] + bv);
      }
  }
}

// --------------------------- flash attention -------------------------------
// Block = 128 thr (4 waves); each wave owns 16 q-rows; block owns 64 q-rows of
// one (b,h). Loops over ALL S keys (masked scores are 1e-9, matching the
// reference's  w*mask + 1e-9*(1-mask)  trick, and there is no 1/sqrt(hd)).
__global__ __launch_bounds__(128) void attn_kernel(
    const _Float16* __restrict__ qkv, _Float16* __restrict__ abuf) {
  constexpr int S = 2048, D = 768, TD = 2304, H = 12;
  __shared__ _Float16 Kt[2][32 * 64];    // [key][d], double buffered
  __shared__ _Float16 Vt[2][32 * 64];
  __shared__ _Float16 Pt[4][16 * 32];    // per-wave P scratch [qrow][key]

  const int tid  = threadIdx.x;
  const int wid  = tid >> 5;
  const int lane = tid & 31;
  const int lo   = lane & 15;
  const int hi   = lane >> 4;

  const int nqb  = S / 64;
  const int bh   = blockIdx.x / nqb;
  const int qblk = blockIdx.x % nqb;
  const int b    = bh / H;
  const int h    = bh % H;
  const int q0   = qblk * 64 + wid * 16;

  const _Float16* Qg = qkv + (size_t)b * S * TD + h * 64;
  const _Float16* Kg = Qg + D;
  const _Float16* Vg = Qg + 2 * D;

  // Load Q fragments (16 rows x 64 K-dim = 2 A-frags), kept in registers.
  V16U qa[2];
  {
    const _Float16* qrow = Qg + (size_t)(q0 + lo) * TD + hi * 16;
#pragma unroll
    for (int kb = 0; kb < 2; ++kb) {
      qa[kb].u[0] = *(const uint4*)(qrow + kb * 32);
      qa[kb].u[1] = *(const uint4*)(qrow + kb * 32 + 8);
    }
  }

  v8f acc[4] = {};          // 16 q-rows x 64 hd cols
  float m_arr[8], l_arr[8];
#pragma unroll
  for (int r = 0; r < 8; ++r) { m_arr[r] = -1e30f; l_arr[r] = 0.0f; }

  auto stage = [&](int j0, int buf) {
    for (int c = tid; c < 256; c += 128) {
      int row = c >> 3, col = (c & 7) * 8;
      cp16(Kg + (size_t)(j0 + row) * TD + col, &Kt[buf][row * 64 + col]);
      cp16(Vg + (size_t)(j0 + row) * TD + col, &Vt[buf][row * 64 + col]);
    }
  };

  stage(0, 0);
  int cur = 0;
  for (int j0 = 0; j0 < S; j0 += 32) {
    cp_wait();           // our async stores for block j0 landed in LDS
    __syncthreads();     // everyone's K/V block j0 visible; prev reads done
    if (j0 + 32 < S) stage(j0 + 32, cur ^ 1);  // prefetch next block

    // ---- scores: batch ALL K-fragment loads, then 4 WMMAs back-to-back ----
    // kf[kb][jj]: kb = K-dim half (0..31 / 32..63), jj = key n-tile (0 / 16)
    V16U kf[2][2];
#pragma unroll
    for (int kb = 0; kb < 2; ++kb) {
      const _Float16* kr0 = &Kt[cur][(0 + lo) * 64 + kb * 32 + hi * 16];
      const _Float16* kr1 = &Kt[cur][(16 + lo) * 64 + kb * 32 + hi * 16];
      kf[kb][0].u[0] = *(const uint4*)(kr0);
      kf[kb][0].u[1] = *(const uint4*)(kr0 + 8);
      kf[kb][1].u[0] = *(const uint4*)(kr1);
      kf[kb][1].u[1] = *(const uint4*)(kr1 + 8);
    }
    v8f c0{}, c1{};
    c0 = wmma16(qa[0].h, kf[0][0].h, c0);
    c1 = wmma16(qa[0].h, kf[0][1].h, c1);
    c0 = wmma16(qa[1].h, kf[1][0].h, c0);
    c1 = wmma16(qa[1].h, kf[1][1].h, c1);

    // mask + online softmax (row statistics via 16-lane-group reductions)
    float sc[8]; _Float16 p0h[8], p1h[8];
    const int qb   = q0 + hi * 8;
    const int key0 = j0 + lo;
    const int key1 = j0 + 16 + lo;
#pragma unroll
    for (int r = 0; r < 8; ++r) {
      int qr = qb + r;
      float s0 = (key0 <= qr) ? c0[r] : 1e-9f;
      float s1 = (key1 <= qr) ? c1[r] : 1e-9f;
      float v = fmaxf(s0, s1);
      v = fmaxf(v, __shfl_xor(v, 1));
      v = fmaxf(v, __shfl_xor(v, 2));
      v = fmaxf(v, __shfl_xor(v, 4));
      v = fmaxf(v, __shfl_xor(v, 8));
      float mn = fmaxf(m_arr[r], v);
      float scale = __expf(m_arr[r] - mn);
      m_arr[r] = mn;
      float p0 = __expf(s0 - mn);
      float p1 = __expf(s1 - mn);
      float rs = p0 + p1;
      rs += __shfl_xor(rs, 1);
      rs += __shfl_xor(rs, 2);
      rs += __shfl_xor(rs, 4);
      rs += __shfl_xor(rs, 8);
      l_arr[r] = l_arr[r] * scale + rs;
      sc[r]  = scale;
      p0h[r] = (_Float16)p0;
      p1h[r] = (_Float16)p1;
    }
#pragma unroll
    for (int t = 0; t < 4; ++t)
#pragma unroll
      for (int r = 0; r < 8; ++r) acc[t][r] *= sc[r];

    // transpose P through per-wave LDS scratch into an A-fragment.
    // (wave-private: LDS ops are in-order within a wave, no barrier needed)
    _Float16* pw = Pt[wid];
#pragma unroll
    for (int r = 0; r < 8; ++r) {
      int prow = hi * 8 + r;
      pw[prow * 32 + lo]      = p0h[r];
      pw[prow * 32 + 16 + lo] = p1h[r];
    }
    V16U pa;
    {
      const _Float16* pr = pw + lo * 32 + hi * 8;
      pa.u[0] = *(const uint4*)(pr);
      pa.u[1] = *(const uint4*)(pr + 16);
    }

    // ---- PV: batch ALL V-fragment gathers, then 4 WMMAs back-to-back ----
    V16U vb[4];
#pragma unroll
    for (int t = 0; t < 4; ++t)
#pragma unroll
      for (int i = 0; i < 16; ++i)
        vb[t].h[i] = Vt[cur][((i >> 3) * 16 + hi * 8 + (i & 7)) * 64 + t * 16 + lo];
#pragma unroll
    for (int t = 0; t < 4; ++t) acc[t] = wmma16(pa.h, vb[t].h, acc[t]);

    cur ^= 1;
  }

  // epilogue: normalize and store merged-heads layout [B,S,D], col = h*64 + d
  _Float16* outp = abuf + (size_t)b * S * D + h * 64;
#pragma unroll
  for (int t = 0; t < 4; ++t)
#pragma unroll
    for (int r = 0; r < 8; ++r) {
      int row = q0 + hi * 8 + r;
      int col = t * 16 + lo;
      outp[(size_t)row * D + col] = (_Float16)(acc[t][r] / l_arr[r]);
    }
}

// ------------------------------ launch -------------------------------------
extern "C" void kernel_launch(void* const* d_in, const int* in_sizes, int n_in,
                              void* d_out, int out_size, void* d_ws,
                              size_t ws_size, hipStream_t stream) {
  (void)in_sizes; (void)n_in; (void)out_size; (void)ws_size;
  const float* x      = (const float*)d_in[0];   // [4,2048,768]
  const float* w_attn = (const float*)d_in[1];   // [768,2304]
  const float* b_attn = (const float*)d_in[2];   // [2304]
  const float* w_proj = (const float*)d_in[3];   // [768,768]
  const float* b_proj = (const float*)d_in[4];   // [768]
  float* out = (float*)d_out;                    // [4,2048,768]

  constexpr int B = 4, S = 2048, D = 768, TD = 3 * D, H = 12;
  const size_t MS = (size_t)B * S;               // 8192 rows

  // workspace carve-out (~68 MB): f16 copies + qkv + attention output
  char* ws = (char*)d_ws;
  size_t off = 0;
  auto carve = [&](size_t bytes) {
    size_t o = off;
    off += (bytes + 255) & ~(size_t)255;
    return o;
  };
  _Float16* xh   = (_Float16*)(ws + carve(MS * D * 2));
  _Float16* wah  = (_Float16*)(ws + carve((size_t)D * TD * 2));
  _Float16* wph  = (_Float16*)(ws + carve((size_t)D * D * 2));
  _Float16* qkvh = (_Float16*)(ws + carve(MS * TD * 2));
  _Float16* abuf = (_Float16*)(ws + carve(MS * D * 2));

  // 1) convert inputs to f16
  {
    int n = (int)(MS * D);
    cvt_f32_f16<<<(n + 255) / 256, 256, 0, stream>>>(x, xh, n);
    n = D * TD;
    cvt_f32_f16<<<(n + 255) / 256, 256, 0, stream>>>(w_attn, wah, n);
    n = D * D;
    cvt_f32_f16<<<(n + 255) / 256, 256, 0, stream>>>(w_proj, wph, n);
  }
  // 2) qkv = x @ w_attn + b_attn   (f16 out)
  gemm_wmma_kernel<_Float16><<<dim3(TD / 128, (int)(MS / 64)), 256, 0, stream>>>(
      xh, wah, b_attn, qkvh, (int)MS, TD, D);
  // 3) flash attention -> abuf (f16, merged heads)
  attn_kernel<<<dim3(B * H * (S / 64)), 128, 0, stream>>>(qkvh, abuf);
  // 4) out = abuf @ w_proj + b_proj (f32 out)
  gemm_wmma_kernel<float><<<dim3(D / 128, (int)(MS / 64)), 256, 0, stream>>>(
      abuf, wph, b_proj, out, (int)MS, D, D);
}
